// GenePropCA_5437428597210
// MI455X (gfx1250) — compile-verified, matrix operand order checked by reading, not run on hardware
//
#include <hip/hip_runtime.h>
#include <hip/hip_bf16.h>

typedef __attribute__((ext_vector_type(16))) _Float16 v16h;
typedef __attribute__((ext_vector_type(8)))  float    v8f;

#define NB   16
#define NC   21
#define NH   256
#define NW   256
#define NPIX (NB * NH * NW)          /* 1,048,576 pixels (b,h,w) */
#define XTOT ((size_t)NB * NC * NH * NW)

/* ---------- 3x3 laplacian, zero padding, kernel LAP/12 ---------- */
__device__ __forceinline__ float lap_zero(const float* __restrict__ pl, int h, int w) {
    float s = -12.f * pl[h * NW + w];
    const bool hm = h > 0, hp = h < NH - 1, wm = w > 0, wp = w < NW - 1;
    if (hm) {
        s += 2.f * pl[(h - 1) * NW + w];
        if (wm) s += pl[(h - 1) * NW + w - 1];
        if (wp) s += pl[(h - 1) * NW + w + 1];
    }
    if (hp) {
        s += 2.f * pl[(h + 1) * NW + w];
        if (wm) s += pl[(h + 1) * NW + w - 1];
        if (wp) s += pl[(h + 1) * NW + w + 1];
    }
    if (wm) s += 2.f * pl[h * NW + w - 1];
    if (wp) s += 2.f * pl[h * NW + w + 1];
    return s * (1.f / 12.f);
}

/* ---------- 3x3 min pool, +inf padding (i.e. min over valid taps) ---------- */
__device__ __forceinline__ float min3_zero(const float* __restrict__ pl, int h, int w) {
    float m = pl[h * NW + w];
    for (int dh = -1; dh <= 1; ++dh) {
        int hh = h + dh;
        if (hh < 0 || hh >= NH) continue;
        for (int dw = -1; dw <= 1; ++dw) {
            int ww = w + dw;
            if (ww < 0 || ww >= NW) continue;
            m = fminf(m, pl[hh * NW + ww]);
        }
    }
    return m;
}

/* =====================================================================
 * Kernel 1: oscillator integration -> new_a/new_b/new_d in workspace
 * ===================================================================== */
__global__ __launch_bounds__(256) void osc1_kernel(
    const float* __restrict__ x,
    const float* __restrict__ slow_w, const float* __restrict__ slow_b,
    const float* __restrict__ pAlpha, const float* __restrict__ pBeta,
    const float* __restrict__ pOmega, const float* __restrict__ pK,
    const float* __restrict__ pKappa,
    float* __restrict__ wa, float* __restrict__ wb, float* __restrict__ wd)
{
    const int p = blockIdx.x * blockDim.x + threadIdx.x;
    if (p >= NPIX) return;
    const int b  = p >> 16;
    const int hw = p & 65535;
    const int h  = hw >> 8;
    const int w  = hw & 255;

    const float alpha = pAlpha[0], beta = pBeta[0], omega = pOmega[0];
    const float Kk = pK[0], kappa = pKappa[0];

    const float* pl3 = x + (((size_t)b * NC + 3) << 16);
    const float alpha_c = pl3[hw];
    const float eroded  = min3_zero(pl3, h, w);
    const float lapA    = lap_zero(pl3, h, w);
    const float q0 = alpha_c;
    const float q1 = alpha_c - eroded;
    const float q2 = lapA;
    const float q3 = x[(((size_t)b * NC + 4) << 16) + hw];
    const float q4 = x[(((size_t)b * NC + 5) << 16) + hw];

    float I[3];
#pragma unroll
    for (int o = 0; o < 3; ++o)
        I[o] = slow_w[o * 5 + 0] * q0 + slow_w[o * 5 + 1] * q1 + slow_w[o * 5 + 2] * q2 +
               slow_w[o * 5 + 3] * q3 + slow_w[o * 5 + 4] * q4 + slow_b[o];

    const float* pa = x + (((size_t)b * NC + 15) << 16);
    const float* pb = x + (((size_t)b * NC + 16) << 16);
    const float* pd = x + (((size_t)b * NC + 17) << 16);
    const float a = pa[hw], bb = pb[hw], d = pd[hw];
    const float la = lap_zero(pa, h, w);
    const float lb = lap_zero(pb, h, w);
    const float ld = lap_zero(pd, h, w);

    wa[p] = a  + 0.1f * (-alpha * a  + omega * bb + Kk * la + I[0]);
    wb[p] = bb + 0.1f * (-alpha * bb - omega * a  + Kk * lb + I[1]);
    wd[p] = d  + 0.1f * (-beta * d + kappa * ld + I[2]);
}

/* =====================================================================
 * Kernel 2: avg-pool5, Kuramoto normalization, phase/amplitude, mod,
 *           prefix copy. Writes out channels 0-11, 15-20, phase, amp.
 * ===================================================================== */
__global__ __launch_bounds__(256) void osc2_kernel(
    const float* __restrict__ x,
    const float* __restrict__ wa, const float* __restrict__ wb,
    const float* __restrict__ wd,
    const float* __restrict__ mod_w, const float* __restrict__ mod_b,
    float* __restrict__ out)
{
    const int p = blockIdx.x * blockDim.x + threadIdx.x;
    if (p >= NPIX) return;
    const int b  = p >> 16;
    const int hw = p & 65535;
    const int h  = hw >> 8;
    const int w  = hw & 255;

    const float* pa = wa + ((size_t)b << 16);
    const float* pb = wb + ((size_t)b << 16);
    float sa = 0.f, sb = 0.f;
    for (int dh = -2; dh <= 2; ++dh) {
        int hh = h + dh;
        if (hh < 0 || hh >= NH) continue;
        for (int dw = -2; dw <= 2; ++dw) {
            int ww = w + dw;
            if (ww < 0 || ww >= NW) continue;
            sa += pa[hh * NW + ww];
            sb += pb[hh * NW + ww];
        }
    }
    sa *= (1.f / 25.f);
    sb *= (1.f / 25.f);
    const float rho = sqrtf(sa * sa + sb * sb + 1e-6f);
    const float aavg = sa / rho, bavg = sb / rho;

    float na = wa[p], nb = wb[p];
    const float nd = wd[p];
    na += 0.1f * (aavg - na);
    nb += 0.1f * (bavg - nb);

    const float phase = sqrtf(na * na + nb * nb + 1e-6f);
    const float amp   = atan2f(nb, na);

    float m[3];
#pragma unroll
    for (int o = 0; o < 3; ++o)
        m[o] = mod_w[o * 3 + 0] * na + mod_w[o * 3 + 1] * nb + mod_w[o * 3 + 2] * nd + mod_b[o];

    const size_t ob = ((size_t)b * NC) << 16;
#pragma unroll
    for (int c = 0; c < 12; ++c)
        out[ob + ((size_t)c << 16) + hw] = x[ob + ((size_t)c << 16) + hw];
    out[ob + ((size_t)15 << 16) + hw] = na;
    out[ob + ((size_t)16 << 16) + hw] = nb;
    out[ob + ((size_t)17 << 16) + hw] = nd;
    out[ob + ((size_t)18 << 16) + hw] = m[0];
    out[ob + ((size_t)19 << 16) + hw] = m[1];
    out[ob + ((size_t)20 << 16) + hw] = m[2];
    out[XTOT + (size_t)p]        = phase;
    out[XTOT + NPIX + (size_t)p] = amp;
}

/* =====================================================================
 * Kernel 3: perception + 84->96->3 MLP on v_wmma_f32_16x16x32_f16.
 * 4 waves/block, 32 pixels/wave (two 16-pixel N-tiles sharing every
 * A-fragment load). LDS holds weights & activations pre-permuted into
 * the 16-bit WMMA A/B fragment layouts (ISA 7.12.2):
 *   element e of lane L maps to K = (e&8?16:0) + (L&16?8:0) + (e&7),
 *   row/col = L&15.
 * Layer-1 completes in registers (12 live v8f accumulators), so the
 * feature buffer is reused for the hidden activations: 45 KB LDS total.
 * ===================================================================== */
#define WAVES 4
__global__ __launch_bounds__(128) void mlp_wmma_kernel(
    const float* __restrict__ x, const float* __restrict__ rnd,
    const float* __restrict__ w1w, const float* __restrict__ w1b,
    const float* __restrict__ w2w, float* __restrict__ out)
{
    __shared__ __align__(32) _Float16 w1LDS[6 * 3 * 512];        /* 18 KB */
    __shared__ __align__(32) _Float16 w2LDS[3 * 512];            /*  3 KB */
    __shared__ __align__(32) _Float16 actLDS[WAVES * 2 * 1536];  /* 24 KB: feat, then hidden */

    const int tid  = threadIdx.x;
    const int lane = tid & 31;
    const int wv   = tid >> 5;

    /* ---- stage layer-1 weights as A-fragments (K padded 84->96) ---- */
    for (int idx = tid; idx < 6 * 3 * 512; idx += 128) {
        const int mm = idx / 1536;
        const int r  = idx - mm * 1536;
        const int t  = r >> 9;
        const int q  = r & 511;
        const int L  = q >> 4;
        const int e  = q & 15;
        const int o  = mm * 16 + (L & 15);
        const int k  = t * 32 + ((e & 8) ? 16 : 0) + ((L & 16) ? 8 : 0) + (e & 7);
        w1LDS[idx] = (_Float16)((k < 84) ? w1w[o * 84 + k] : 0.f);
    }
    /* ---- stage layer-2 weights (M padded 3->16, K = 96) ---- */
    for (int idx = tid; idx < 3 * 512; idx += 128) {
        const int t = idx >> 9;
        const int q = idx & 511;
        const int L = q >> 4;
        const int e = q & 15;
        const int M = L & 15;
        const int k = t * 32 + ((e & 8) ? 16 : 0) + ((L & 16) ? 8 : 0) + (e & 7);
        w2LDS[idx] = (_Float16)((M < 3) ? w2w[M * 96 + k] : 0.f);
    }

    /* ---- this wave's 32 pixels (consecutive w within one row) ---- */
    const int pixBase = blockIdx.x * (WAVES * 32) + wv * 32;
    const int b   = pixBase >> 16;
    const int hw0 = pixBase & 65535;
    const int h   = hw0 >> 8;
    const int w0  = hw0 & 255;
    const int n   = lane & 15;           /* column within an N-tile      */
    const int gw  = w0 + lane;           /* this lane's own pixel column */

    const int hm = (h + NH - 1) & (NH - 1), hp = (h + 1) & (NH - 1);
    const int wm = (gw + NW - 1) & (NW - 1), wp = (gw + 1) & (NW - 1);

    _Float16* act  = actLDS + wv * 3072;        /* two 1536-half tiles  */
    _Float16* mine = act + (lane >> 4) * 1536;  /* this lane's N-tile   */

    /* zero-pad K rows 84..95 of the B fragments (each lane: own (tile,n)) */
#pragma unroll
    for (int k = 84; k < 96; ++k) {
        const int t  = k >> 5, kk = k & 31;
        const int Ld = n + ((kk & 8) ? 16 : 0);
        const int e  = ((kk & 16) ? 8 : 0) + (kk & 7);
        mine[(t * 32 + Ld) * 16 + e] = (_Float16)0.f;
    }

    /* ---- perception features (wrap padding), scattered as B-fragments.
           Each lane owns one pixel and walks all 21 channels. ---- */
    for (int c = 0; c < NC; ++c) {
        const float* pl = x + (((size_t)b * NC + c) << 16);
        if (c + 1 < NC) {   /* hide HBM latency for the next plane */
            const float* nx = x + (((size_t)b * NC + c + 1) << 16);
            __builtin_prefetch(nx + h * NW + gw, 0, 1);
        }
        const float x00 = pl[hm * NW + wm], x01 = pl[hm * NW + gw], x02 = pl[hm * NW + wp];
        const float x10 = pl[h  * NW + wm], x11 = pl[h  * NW + gw], x12 = pl[h  * NW + wp];
        const float x20 = pl[hp * NW + wm], x21 = pl[hp * NW + gw], x22 = pl[hp * NW + wp];
        const float sx = (x02 - x00) + 2.f * (x12 - x10) + (x22 - x20);
        const float sy = (x20 - x00) + 2.f * (x21 - x01) + (x22 - x02);
        const float lp = x00 + x02 + x20 + x22 + 2.f * (x01 + x10 + x12 + x21) - 12.f * x11;
        const int   ks[4] = { c, 21 + 3 * c + 0, 21 + 3 * c + 1, 21 + 3 * c + 2 };
        const float vs[4] = { x11, sx, sy, lp };
#pragma unroll
        for (int j = 0; j < 4; ++j) {
            const int k  = ks[j];
            const int t  = k >> 5, kk = k & 31;
            const int Ld = n + ((kk & 8) ? 16 : 0);
            const int e  = ((kk & 16) ? 8 : 0) + (kk & 7);
            mine[(t * 32 + Ld) * 16 + e] = (_Float16)vs[j];
        }
    }

    __syncthreads();

    const int hi8 = (lane >= 16) ? 8 : 0;

    /* ---- layer 1: 6 M-tiles x 3 K-steps x 2 N-tiles; A shared per pair ---- */
    v8f acc[6][2];
#pragma unroll
    for (int mm = 0; mm < 6; ++mm) {
#pragma unroll
        for (int r = 0; r < 8; ++r) {
            const float bv = w1b[mm * 16 + r + hi8];
            acc[mm][0][r] = bv;
            acc[mm][1][r] = bv;
        }
#pragma unroll
        for (int t = 0; t < 3; ++t) {
            v16h afrag = *(const v16h*)&w1LDS[(mm * 3 + t) * 512 + lane * 16];
            v16h b0    = *(const v16h*)&act[t * 512 + lane * 16];
            v16h b1    = *(const v16h*)&act[1536 + t * 512 + lane * 16];
            acc[mm][0] = __builtin_amdgcn_wmma_f32_16x16x32_f16(
                false, afrag, false, b0, (short)0, acc[mm][0], false, false);
            acc[mm][1] = __builtin_amdgcn_wmma_f32_16x16x32_f16(
                false, afrag, false, b1, (short)0, acc[mm][1], false, false);
        }
    }

    __syncthreads();   /* feature reads done everywhere; reuse actLDS */

    /* ---- ReLU + scatter hidden as layer-2 B-fragments (into actLDS) ---- */
#pragma unroll
    for (int mm = 0; mm < 6; ++mm) {
#pragma unroll
        for (int r = 0; r < 8; ++r) {
            const int o  = mm * 16 + r + hi8;
            const int t2 = o >> 5, kk = o & 31;
            const int Ld = n + ((kk & 8) ? 16 : 0);
            const int e  = ((kk & 16) ? 8 : 0) + (kk & 7);
            const float v0 = acc[mm][0][r] > 0.f ? acc[mm][0][r] : 0.f;
            const float v1 = acc[mm][1][r] > 0.f ? acc[mm][1][r] : 0.f;
            act[(t2 * 32 + Ld) * 16 + e]        = (_Float16)v0;
            act[1536 + (t2 * 32 + Ld) * 16 + e] = (_Float16)v1;
        }
    }

    __syncthreads();

    /* ---- layer 2: one M-tile (rows 0..2 valid), 3 K-steps x 2 N-tiles ---- */
    v8f acc2[2];
#pragma unroll
    for (int s = 0; s < 2; ++s)
#pragma unroll
        for (int r = 0; r < 8; ++r) acc2[s][r] = 0.f;
#pragma unroll
    for (int t = 0; t < 3; ++t) {
        v16h afrag = *(const v16h*)&w2LDS[t * 512 + lane * 16];
        v16h b0    = *(const v16h*)&act[t * 512 + lane * 16];
        v16h b1    = *(const v16h*)&act[1536 + t * 512 + lane * 16];
        acc2[0] = __builtin_amdgcn_wmma_f32_16x16x32_f16(
            false, afrag, false, b0, (short)0, acc2[0], false, false);
        acc2[1] = __builtin_amdgcn_wmma_f32_16x16x32_f16(
            false, afrag, false, b1, (short)0, acc2[1], false, false);
    }

    /* ---- gene update (rows 0..2 live in lanes 0..15, VGPRs 0..2) ---- */
    if (lane < 16) {
        const float* pl3 = x + (((size_t)b * NC + 3) << 16);
#pragma unroll
        for (int s = 0; s < 2; ++s) {
            const int gws = w0 + s * 16 + lane;
            const int wms = (gws + NW - 1) & (NW - 1), wps = (gws + 1) & (NW - 1);
            const size_t pix = ((size_t)b << 16) + (size_t)(h * NW + gws);
            const float mask = floorf(rnd[pix] + 0.5f);
            float mx = pl3[hm * NW + wms];
            mx = fmaxf(mx, pl3[hm * NW + gws]);
            mx = fmaxf(mx, pl3[hm * NW + wps]);
            mx = fmaxf(mx, pl3[h  * NW + wms]);
            mx = fmaxf(mx, pl3[h  * NW + gws]);
            mx = fmaxf(mx, pl3[h  * NW + wps]);
            mx = fmaxf(mx, pl3[hp * NW + wms]);
            mx = fmaxf(mx, pl3[hp * NW + gws]);
            mx = fmaxf(mx, pl3[hp * NW + wps]);
            const float life = (mx > 0.1f) ? 1.f : 0.f;
            const float sc = mask * life;
#pragma unroll
            for (int g = 0; g < 3; ++g) {
                const size_t oi = (((size_t)b * NC + 12 + g) << 16) + (size_t)(h * NW + gws);
                out[oi] = x[oi] + acc2[s][g] * sc;
            }
        }
    }
}

/* ===================================================================== */
extern "C" void kernel_launch(void* const* d_in, const int* in_sizes, int n_in,
                              void* d_out, int out_size, void* d_ws, size_t ws_size,
                              hipStream_t stream) {
    (void)in_sizes; (void)n_in; (void)out_size; (void)ws_size;
    const float* x      = (const float*)d_in[0];
    const float* rnd    = (const float*)d_in[1];
    const float* w1w    = (const float*)d_in[2];
    const float* w1b    = (const float*)d_in[3];
    const float* w2w    = (const float*)d_in[4];
    const float* slw    = (const float*)d_in[5];
    const float* slb    = (const float*)d_in[6];
    const float* mow    = (const float*)d_in[7];
    const float* mob    = (const float*)d_in[8];
    const float* pAlpha = (const float*)d_in[9];
    const float* pBeta  = (const float*)d_in[10];
    const float* pOmega = (const float*)d_in[11];
    const float* pK     = (const float*)d_in[12];
    const float* pKappa = (const float*)d_in[13];
    float* out = (float*)d_out;

    float* wa = (float*)d_ws;
    float* wb = wa + NPIX;
    float* wd = wb + NPIX;

    osc1_kernel<<<NPIX / 256, 256, 0, stream>>>(x, slw, slb, pAlpha, pBeta,
                                                pOmega, pK, pKappa, wa, wb, wd);
    osc2_kernel<<<NPIX / 256, 256, 0, stream>>>(x, wa, wb, wd, mow, mob, out);
    mlp_wmma_kernel<<<NPIX / (WAVES * 32), WAVES * 32, 0, stream>>>(x, rnd, w1w, w1b, w2w, out);
}
